// NeuralCDE_25555055411506
// MI455X (gfx1250) — matrix operand order repaired and checked by reading, not compiled, not run on hardware
//
#include <hip/hip_runtime.h>

#define SB   1024   // S
#define BB   1024   // B
#define CC   8      // channels
#define HH   32     // hidden
#define HIDN 128    // CDEFunc inner width
#define NINT (SB - 2)

typedef __attribute__((ext_vector_type(16))) __bf16       v16bf;
typedef __attribute__((ext_vector_type(2)))  __bf16       v2bf;
typedef __attribute__((ext_vector_type(8)))  float        v8f;
typedef __attribute__((ext_vector_type(8)))  unsigned int v8u;

union BFU { v8u u; v16bf v; };

// pack two floats into bf16x2 — gfx1250 has native v_cvt_pk_bf16_f32
__device__ __forceinline__ unsigned int pack2bf(float lo, float hi) {
#if __has_builtin(__builtin_amdgcn_cvt_pk_bf16_f32)
    v2bf r = __builtin_amdgcn_cvt_pk_bf16_f32(lo, hi);
    return __builtin_bit_cast(unsigned int, r);
#else
    unsigned int r;
    asm("v_cvt_pk_bf16_f32 %0, %1, %2" : "=v"(r) : "v"(lo), "v"(hi));
    return r;
#endif
}
// single-instruction ReLU: v_med3_f32(x, 0, +inf)
__device__ __forceinline__ float relu(float x) {
#if __has_builtin(__builtin_amdgcn_fmed3f)
    return __builtin_amdgcn_fmed3f(x, 0.0f, __builtin_inff());
#else
    return fmaxf(x, 0.0f);
#endif
}
// hardware tanh when available (gfx1250 trans op), else exp-based (inf-safe)
#if __has_builtin(__builtin_amdgcn_tanhf)
__device__ __forceinline__ float fast_tanh(float x) { return __builtin_amdgcn_tanhf(x); }
#elif __has_builtin(__builtin_amdgcn_tanh_f32)
__device__ __forceinline__ float fast_tanh(float x) { return __builtin_amdgcn_tanh_f32(x); }
#else
__device__ __forceinline__ float fast_tanh(float x) {
    float e = __expf(2.0f * x);
    return 1.0f - 2.0f / (e + 1.0f);
}
#endif

__device__ __forceinline__ v8f wmma_bf16(v16bf a, v16bf b, v8f c) {
    return __builtin_amdgcn_wmma_f32_16x16x32_bf16(false, a, false, b, (short)0, c,
                                                   false, false);
}
// Build a 16x32 bf16 A-fragment / 32x16 B-fragment from two contiguous 16B LDS chunks.
__device__ __forceinline__ v16bf mkfrag_lds(const unsigned short* p, int off0, int off1) {
    uint4 a = *(const uint4*)(p + off0);
    uint4 b = *(const uint4*)(p + off1);
    BFU t;
    t.u[0] = a.x; t.u[1] = a.y; t.u[2] = a.z; t.u[3] = a.w;
    t.u[4] = b.x; t.u[5] = b.y; t.u[6] = b.z; t.u[7] = b.w;
    return t.v;
}

// ---------------------------------------------------------------------------
// Kernel 1: shared Thomas-algorithm coefficients c'_i = 1/(4 - c'_{i-1})
// ---------------------------------------------------------------------------
__global__ void cprime_kernel(float* __restrict__ cp) {
    if (blockIdx.x == 0 && threadIdx.x == 0) {
        float c = 0.25f;
        cp[0] = c;
        for (int i = 1; i < NINT; ++i) { c = 1.0f / (4.0f - c); cp[i] = c; }
    }
}

// ---------------------------------------------------------------------------
// Kernel 2: natural cubic spline second derivatives, one thread per (b,c)
// column. tridiag(1,4,1) m = 6*S^2 * second-difference(X).  d' stored into M
// in place, backward substitution overwrites with the solution.
// M layout: (B, S, C) f32.
// ---------------------------------------------------------------------------
__global__ __launch_bounds__(128) void spline_kernel(const float* __restrict__ X,
                                                     const float* __restrict__ cp,
                                                     float* __restrict__ M) {
    const int t = blockIdx.x * blockDim.x + threadIdx.x;   // 0..8191
    const int b = t >> 3, c = t & 7;
    const float* xp = X + (size_t)b * (SB * CC) + c;
    float*       mp = M + (size_t)b * (SB * CC) + c;
    const float  K  = 6.0f * (float)SB * (float)SB;

    mp[0] = 0.0f;
    mp[(size_t)(SB - 1) * CC] = 0.0f;

    float x0 = xp[0], x1 = xp[CC];
    float d = 0.0f;
    for (int i = 1; i <= NINT; ++i) {            // forward sweep
        float x2 = xp[(size_t)(i + 1) * CC];
        float r  = K * (x2 - 2.0f * x1 + x0);
        d = (r - d) * cp[i - 1];
        mp[(size_t)i * CC] = d;
        x0 = x1; x1 = x2;
    }
    float m = 0.0f;
    for (int j = NINT; j >= 1; --j) {            // backward substitution
        float dj = mp[(size_t)j * CC];
        m = dj - cp[j - 1] * m;
        mp[(size_t)j * CC] = m;
    }
}

// ---------------------------------------------------------------------------
// Kernel 3: persistent RK4 CDE scan.  64 blocks x 128 threads; each block owns
// 16 batch rows.  Wave w owns u-columns [32w,32w+32) and v-columns [64w,64w+64).
// Weights are pre-fragmented into registers (bf16 B-operands) before the loop.
// ---------------------------------------------------------------------------
__global__ __launch_bounds__(128) void cde_kernel(
    const float* __restrict__ X,  const float* __restrict__ M,
    const float* __restrict__ Wi, const float* __restrict__ bi,
    const float* __restrict__ W1, const float* __restrict__ b1,
    const float* __restrict__ W2, const float* __restrict__ b2,
    const float* __restrict__ Wr, const float* __restrict__ br,
    float* __restrict__ out) {

    __shared__ unsigned short zbf[16 * HH];      // stage z, bf16, row-major
    __shared__ unsigned short ubf[16 * HIDN];    // relu(zW1+b1), bf16
    __shared__ float          vsh[16 * 256];     // tanh(uW2+b2), f32
    __shared__ float          dxsh[3 * 128];     // dX at tau = {0, h/2, h}
    __shared__ float          ztmp[16 * HH];     // final z for readout

    const int tid = threadIdx.x;
    const int w   = tid >> 5;          // wave 0..3
    const int l   = tid & 31;
    const int lm  = l & 15;
    const int lh  = (l >> 4) & 1;      // lane half
    const int bg  = blockIdx.x * 16;   // batch base

    const float hstep = 1.0f / (float)SB;
    const float invh  = (float)SB;

    // per-thread RK4 state: row rb, columns hb..hb+3
    const int rb = tid >> 3;
    const int cc = tid & 7;
    const int hb = cc * 4;

    float zr[4], ks[4];
    {   // z0 = X[:,0,:] @ W_init + b_init
        float x0[CC];
#pragma unroll
        for (int c = 0; c < CC; ++c) x0[c] = X[(size_t)(bg + rb) * (SB * CC) + c];
#pragma unroll
        for (int j = 0; j < 4; ++j) {
            float a = bi[hb + j];
#pragma unroll
            for (int c = 0; c < CC; ++c) a += x0[c] * Wi[c * HH + hb + j];
            zr[j] = a; ks[j] = 0.0f;
        }
        uint2 pz; pz.x = pack2bf(zr[0], zr[1]); pz.y = pack2bf(zr[2], zr[3]);
        *(uint2*)&zbf[rb * HH + hb] = pz;
    }

    // --- weight B-fragments into registers (held across all 1023 steps) ---
    v16bf w1f[2]; float bias1[2];
#pragma unroll
    for (int tt = 0; tt < 2; ++tt) {
        const int n  = (w * 2 + tt) * 16 + lm;
        const int kb = lh ? 16 : 0;
        BFU t;
#pragma unroll
        for (int j = 0; j < 8; ++j)
            t.u[j] = pack2bf(W1[(kb + 2 * j) * HIDN + n], W1[(kb + 2 * j + 1) * HIDN + n]);
        w1f[tt] = t.v; bias1[tt] = b1[n];
    }
    v16bf w2f[4][4]; float bias2[4];
#pragma unroll
    for (int tt = 0; tt < 4; ++tt) {
        const int n = (w * 4 + tt) * 16 + lm;
        bias2[tt] = b2[n];
#pragma unroll
        for (int kc = 0; kc < 4; ++kc) {
            const int kb = kc * 32 + (lh ? 16 : 0);
            BFU t;
#pragma unroll
            for (int j = 0; j < 8; ++j)
                t.u[j] = pack2bf(W2[(kb + 2 * j) * 256 + n], W2[(kb + 2 * j + 1) * 256 + n]);
            w2f[tt][kc] = t.v;
        }
    }
    __syncthreads();

    // ------------------------------ time scan ------------------------------
    for (int s = 0; s < SB - 1; ++s) {
        {   // spline derivative dX(tau) for this step; k2/k3 share tau=h/2
            const size_t xi = (size_t)(bg + rb) * (SB * CC) + (size_t)s * CC + cc;
            float y0 = X[xi], y1 = X[xi + CC];
            float m0 = M[xi], m1 = M[xi + CC];
            __builtin_prefetch(&X[xi + 2 * CC], 0, 3);
            __builtin_prefetch(&M[xi + 2 * CC], 0, 3);
            float base = (y1 - y0) * invh - (m1 - m0) * (hstep * (1.0f / 6.0f));
            dxsh[0 * 128 + tid] = base - m0 * (0.5f * hstep);
            dxsh[1 * 128 + tid] = base + (m1 - m0) * (0.125f * hstep);
            dxsh[2 * 128 + tid] = base + m1 * (0.5f * hstep);
        }
#pragma unroll
        for (int st = 0; st < 4; ++st) {
            {   // phase A: u = relu(z @ W1 + b1), this wave's 32 columns
                const int akb = lh ? 16 : 0;
                v16bf zf = mkfrag_lds(zbf, lm * HH + akb, lm * HH + akb + 8);
#pragma unroll
                for (int tt = 0; tt < 2; ++tt) {
                    v8f acc;
#pragma unroll
                    for (int j = 0; j < 8; ++j) acc[j] = bias1[tt];
                    acc = wmma_bf16(zf, w1f[tt], acc);
                    const int n = (w * 2 + tt) * 16 + lm;
#pragma unroll
                    for (int j = 0; j < 8; j += 2) {   // v_med3 relu + v_cvt_pk_bf16
                        unsigned int p = pack2bf(relu(acc[j]), relu(acc[j + 1]));
                        ubf[(j + 0 + lh * 8) * HIDN + n] = (unsigned short)p;
                        ubf[(j + 1 + lh * 8) * HIDN + n] = (unsigned short)(p >> 16);
                    }
                }
            }
            __syncthreads();
            {   // phase B: v = tanh(u @ W2 + b2), this wave's 64 columns
                v8f acc2[4];
#pragma unroll
                for (int tt = 0; tt < 4; ++tt)
#pragma unroll
                    for (int j = 0; j < 8; ++j) acc2[tt][j] = bias2[tt];
#pragma unroll
                for (int kc = 0; kc < 4; ++kc) {
                    const int kb = kc * 32 + (lh ? 16 : 0);
                    v16bf uf = mkfrag_lds(ubf, lm * HIDN + kb, lm * HIDN + kb + 8);
#pragma unroll
                    for (int tt = 0; tt < 4; ++tt)
                        acc2[tt] = wmma_bf16(uf, w2f[tt][kc], acc2[tt]);
                }
#pragma unroll
                for (int tt = 0; tt < 4; ++tt) {
                    const int n = (w * 4 + tt) * 16 + lm;
#pragma unroll
                    for (int j = 0; j < 8; ++j)
                        vsh[(j + lh * 8) * 256 + n] = fast_tanh(acc2[tt][j]);
                }
            }
            __syncthreads();
            {   // phase C: k[b,h] = sum_c v[b,h*8+c]*dX[b,c]; RK4 combine
                const int dsel = (st == 0) ? 0 : (st == 3) ? 2 : 1;
                float dx[8];
#pragma unroll
                for (int c = 0; c < CC; ++c) dx[c] = dxsh[dsel * 128 + rb * 8 + c];
                float kv[4];
#pragma unroll
                for (int j = 0; j < 4; ++j) {
                    const float* vp = &vsh[rb * 256 + (hb + j) * 8];
                    float a = 0.0f;
#pragma unroll
                    for (int c = 0; c < CC; ++c) a += vp[c] * dx[c];
                    kv[j] = a;
                }
                float zn[4];
                if (st < 3) {
                    const float kw = (st == 0) ? 1.0f : 2.0f;
                    const float cs = (st == 2) ? hstep : 0.5f * hstep;
#pragma unroll
                    for (int j = 0; j < 4; ++j) {
                        ks[j] += kw * kv[j];
                        zn[j] = zr[j] + cs * kv[j];
                    }
                } else {
#pragma unroll
                    for (int j = 0; j < 4; ++j) {
                        ks[j] += kv[j];
                        zr[j] += (hstep * (1.0f / 6.0f)) * ks[j];
                        ks[j] = 0.0f;
                        zn[j] = zr[j];
                    }
                }
                uint2 pz; pz.x = pack2bf(zn[0], zn[1]); pz.y = pack2bf(zn[2], zn[3]);
                *(uint2*)&zbf[rb * HH + hb] = pz;
            }
            __syncthreads();
        }
    }

    // ------------------------------ readout ------------------------------
#pragma unroll
    for (int j = 0; j < 4; ++j) ztmp[rb * HH + hb + j] = zr[j];
    __syncthreads();
    if (tid < 160) {
        const int bo = tid / 10, o = tid % 10;
        float a = br[o];
#pragma unroll
        for (int hh = 0; hh < HH; ++hh) a += ztmp[bo * HH + hh] * Wr[hh * 10 + o];
        out[(size_t)(bg + bo) * 10 + o] = a;
    }
}

// ---------------------------------------------------------------------------
extern "C" void kernel_launch(void* const* d_in, const int* in_sizes, int n_in,
                              void* d_out, int out_size, void* d_ws, size_t ws_size,
                              hipStream_t stream) {
    (void)in_sizes; (void)n_in; (void)out_size; (void)ws_size;
    const float* X  = (const float*)d_in[1];   // (B,S,C)
    const float* Wi = (const float*)d_in[2];   // (C,H)
    const float* bi = (const float*)d_in[3];   // (H)
    const float* W1 = (const float*)d_in[4];   // (H,HID)
    const float* b1 = (const float*)d_in[5];   // (HID)
    const float* W2 = (const float*)d_in[6];   // (HID,C*H)
    const float* b2 = (const float*)d_in[7];   // (C*H)
    const float* Wr = (const float*)d_in[8];   // (H,O)
    const float* br = (const float*)d_in[9];   // (O)
    float* out = (float*)d_out;

    float* M  = (float*)d_ws;                          // (B,S,C) = 32 MB
    float* cp = M + (size_t)BB * SB * CC;              // 1022 floats

    cprime_kernel<<<1, 1, 0, stream>>>(cp);
    spline_kernel<<<(BB * CC) / 128, 128, 0, stream>>>(X, cp, M);
    cde_kernel<<<BB / 16, 128, 0, stream>>>(X, M, Wi, bi, W1, b1, W2, b2, Wr, br, out);
}
